// LocClusterNet_53008486367324
// MI455X (gfx1250) — matrix operand order repaired and checked by reading, not compiled, not run on hardware
//
#include <hip/hip_runtime.h>
#include <cstdint>
#include <cstddef>

// ---------------------------------------------------------------------------
// CDNA5 (gfx1250) implementation of LocClusterNet forward pass.
//  - Edge MLP + scatter-max phase: fp32 VALU, LDS-staged folded weights,
//    uint-bit atomicMax (relu commutes with max; L2-resident gathers).
//  - SAGEConv GEMMs: v_wmma_f32_16x16x32_f16 tiles (16 rows, K padded to 32).
// ---------------------------------------------------------------------------

typedef __attribute__((ext_vector_type(16))) _Float16 v16h;
typedef __attribute__((ext_vector_type(8)))  float    v8f;

#define BN_SCALE_F 0.9999950000375f  // 1/sqrt(1 + 1e-5)

// ---------------------------------------------------------------------------
// generic fill
// ---------------------------------------------------------------------------
__global__ void fill_u32_kernel(unsigned* __restrict__ p, unsigned v, long n) {
  long i = (long)blockIdx.x * blockDim.x + threadIdx.x;
  if (i < n) p[i] = v;
}

// ---------------------------------------------------------------------------
// fold BatchNorm (eval mode) into the MLP linear layers
// layout per mlp m (float offsets 0 / 60 / 186): [W0 bD][b0][W1][b1][W2][b2]
// ---------------------------------------------------------------------------
struct FoldArgs {
  const float* W[9];
  const float* B[9];
  const float* G[6];
  const float* BNB[6];
  float*       out;
};

__global__ void fold_kernel(FoldArgs a) {
  const int dims[3] = {4, 6, 8};
  const int offs[3] = {0, 60, 186};
  int tid = threadIdx.x;
  if (tid >= 9) return;
  int m = tid / 3, l = tid % 3;
  int D = dims[m];
  float* Wo = a.out + offs[m] + l * (D * D + D);
  float* bo = Wo + D * D;
  const float* W = a.W[m * 3 + l];
  const float* b = a.B[m * 3 + l];
  if (l < 2) {
    const float* g  = a.G[m * 2 + l];
    const float* bb = a.BNB[m * 2 + l];
    for (int j = 0; j < D; ++j) {
      float sc = g[j] * BN_SCALE_F;
      for (int k = 0; k < D; ++k) Wo[k * D + j] = W[k * D + j] * sc;
      bo[j] = b[j] * sc + bb[j];
    }
  } else {
    for (int i = 0; i < D * D; ++i) Wo[i] = W[i];
    for (int j = 0; j < D; ++j) bo[j] = b[j];
  }
}

// ---------------------------------------------------------------------------
// PointNetConv edge kernel: m = MLP(cat(x[src], pos[src]-pos[dst]))
// then atomicMax(relu(m)) into out[dst].  out zero-initialized; values >= 0
// so uint-bit compare == float compare.
// ---------------------------------------------------------------------------
template <int DPREV>
__global__ void edge_mlp_kernel(const int* __restrict__ src, const int* __restrict__ dst,
                                const float* __restrict__ pos, const float* __restrict__ xin,
                                const float* __restrict__ wf, float* __restrict__ out, int nE) {
  constexpr int D   = DPREV + 2;
  constexpr int LSZ = 3 * (D * D + D);
  __shared__ float w[LSZ];
  for (int i = threadIdx.x; i < LSZ; i += blockDim.x) w[i] = wf[i];
  __syncthreads();
  int e = blockIdx.x * blockDim.x + threadIdx.x;
  if (e >= nE) return;
  int s = src[e];
  int t = dst[e];
  const float* W0 = w;
  const float* b0 = W0 + D * D;
  const float* W1 = b0 + D;
  const float* b1 = W1 + D * D;
  const float* W2 = b1 + D;
  const float* b2 = W2 + D * D;
  float h0[D], h1[D];
#pragma unroll
  for (int k = 0; k < DPREV; ++k) h0[k] = xin[s * DPREV + k];
  h0[DPREV]     = pos[2 * s]     - pos[2 * t];
  h0[DPREV + 1] = pos[2 * s + 1] - pos[2 * t + 1];
#pragma unroll
  for (int j = 0; j < D; ++j) {
    float a = b0[j];
#pragma unroll
    for (int k = 0; k < D; ++k) a += h0[k] * W0[k * D + j];
    h1[j] = fmaxf(a, 0.0f);
  }
#pragma unroll
  for (int j = 0; j < D; ++j) {
    float a = b1[j];
#pragma unroll
    for (int k = 0; k < D; ++k) a += h1[k] * W1[k * D + j];
    h0[j] = fmaxf(a, 0.0f);
  }
  unsigned* o = (unsigned*)(out + (size_t)t * D);
#pragma unroll
  for (int j = 0; j < D; ++j) {
    float a = b2[j];
#pragma unroll
    for (int k = 0; k < D; ++k) a += h0[k] * W2[k * D + j];
    a = fmaxf(a, 0.0f);  // relu commutes with segment-max (monotonic)
    atomicMax(o + j, __float_as_uint(a));
  }
}

// ---------------------------------------------------------------------------
// Loc->Cluster max aggregation (x3 >= 0 post-relu; zero-init target)
// ---------------------------------------------------------------------------
__global__ void cagg_kernel(const int* __restrict__ lcsrc, const int* __restrict__ lcdst,
                            const float* __restrict__ x3, float* __restrict__ cagg, int n) {
  int i = blockIdx.x * blockDim.x + threadIdx.x;
  if (i >= n) return;
  int s = lcsrc[i];
  int c = lcdst[i];
  unsigned* o = (unsigned*)(cagg + (size_t)c * 8);
#pragma unroll
  for (int j = 0; j < 8; ++j) atomicMax(o + j, __float_as_uint(x3[(size_t)s * 8 + j]));
}

__global__ void concat_kernel(const float* __restrict__ xcl, const float* __restrict__ cagg,
                              float* __restrict__ xc0, int nCl) {
  int i = blockIdx.x * blockDim.x + threadIdx.x;
  if (i >= nCl * 17) return;
  int c = i / 17, j = i % 17;
  xc0[i] = (j < 9) ? xcl[c * 9 + j] : cagg[c * 8 + (j - 9)];
}

// ---------------------------------------------------------------------------
// segment count + sum for SAGE mean aggregation
// ---------------------------------------------------------------------------
__global__ void count_kernel(const int* __restrict__ t, float* __restrict__ cnt, int n) {
  int e = blockIdx.x * blockDim.x + threadIdx.x;
  if (e < n) atomicAdd(&cnt[t[e]], 1.0f);
}

__global__ void seg_sum_kernel(const int* __restrict__ s, const int* __restrict__ t,
                               const float* __restrict__ xc, float* __restrict__ sum,
                               int d, int n) {
  int e = blockIdx.x * blockDim.x + threadIdx.x;
  if (e >= n) return;
  int a = s[e], b = t[e];
  for (int j = 0; j < d; ++j) atomicAdd(&sum[(size_t)b * d + j], xc[(size_t)a * d + j]);
}

// ---------------------------------------------------------------------------
// SAGEConv:  out = (sum/max(cnt,1)) @ Wl + bl + xc @ Wr       via WMMA
// One wave handles a 16-row tile; K padded to 32, DOUT split into 16-col tiles.
// A layout (f16 16x32): lane half h picks K chunks {8h..8h+7, 16+8h..16+8h+7}
// B layout (f16 32x16): lane = column, same K chunks as A {8h..8h+7, 16+8h..16+8h+7}
// D layout (f32 16x16): VGPR j -> M = 8h + j, lane -> column
// ---------------------------------------------------------------------------
template <int DIN, int DOUT>
__global__ void sage_gemm_kernel(const float* __restrict__ sum, const float* __restrict__ cnt,
                                 const float* __restrict__ xc,
                                 const float* __restrict__ Wl, const float* __restrict__ bl,
                                 const float* __restrict__ Wr,
                                 float* __restrict__ out, int nCl) {
  int lane   = threadIdx.x & 31;
  int tile   = blockIdx.x * (blockDim.x >> 5) + (threadIdx.x >> 5);
  int nTiles = (nCl + 15) >> 4;
  if (tile >= nTiles) return;  // uniform per wave: EXEC stays all-ones for WMMA
  int  h     = lane >> 4;
  int  lm    = lane & 15;
  int  row   = tile * 16 + lm;
  bool rowOk = row < nCl;
  float invc = rowOk ? (1.0f / fmaxf(cnt[row], 1.0f)) : 1.0f;

  v16h aM, aX;
#pragma unroll
  for (int i = 0; i < 16; ++i) {
    int   k  = ((i >> 3) << 4) + (h << 3) + (i & 7);
    float vm = 0.0f, vx = 0.0f;
    if (rowOk && k < DIN) {
      vm = sum[(size_t)row * DIN + k] * invc;
      vx = xc[(size_t)row * DIN + k];
    }
    aM[i] = (_Float16)vm;
    aX[i] = (_Float16)vx;
  }

  constexpr int NT = (DOUT + 15) / 16;
#pragma unroll
  for (int nt = 0; nt < NT; ++nt) {
    int  col = nt * 16 + lm;
    v16h bL, bR;
#pragma unroll
    for (int i = 0; i < 16; ++i) {
      int   k  = ((i >> 3) << 4) + (h << 3) + (i & 7);
      float wl = 0.0f, wr = 0.0f;
      if (k < DIN && col < DOUT) {
        wl = Wl[k * DOUT + col];
        wr = Wr[k * DOUT + col];
      }
      bL[i] = (_Float16)wl;
      bR[i] = (_Float16)wr;
    }
    float bias = (col < DOUT) ? bl[col] : 0.0f;
    v8f acc;
#pragma unroll
    for (int j = 0; j < 8; ++j) acc[j] = bias;
    acc = __builtin_amdgcn_wmma_f32_16x16x32_f16(false, aM, false, bL, (short)0, acc, false, false);
    acc = __builtin_amdgcn_wmma_f32_16x16x32_f16(false, aX, false, bR, (short)0, acc, false, false);
    if (col < DOUT) {
#pragma unroll
      for (int j = 0; j < 8; ++j) {
        int r = tile * 16 + (h << 3) + j;
        if (r < nCl) out[(size_t)r * DOUT + col] = acc[j];
      }
    }
  }
}

// ---------------------------------------------------------------------------
// graph max-pool with order-preserving uint key (handles negatives, -inf init)
// ---------------------------------------------------------------------------
__device__ __forceinline__ unsigned pool_enc(float f) {
  unsigned b = __float_as_uint(f);
  return (b & 0x80000000u) ? ~b : (b | 0x80000000u);
}
__device__ __forceinline__ float pool_dec(unsigned k) {
  unsigned b = (k & 0x80000000u) ? (k & 0x7fffffffu) : ~k;
  return __uint_as_float(b);
}
#define POOL_NEG_INF_KEY 0x007FFFFFu  // pool_enc(-inf)

__global__ void pool_kernel(const int* __restrict__ batch, const float* __restrict__ xc,
                            unsigned* __restrict__ pooled, int nCl) {
  int c = blockIdx.x * blockDim.x + threadIdx.x;
  if (c >= nCl) return;
  int g = batch[c];
#pragma unroll
  for (int j = 0; j < 32; ++j) atomicMax(&pooled[(size_t)g * 32 + j], pool_enc(xc[(size_t)c * 32 + j]));
}

__global__ void head_kernel(const unsigned* __restrict__ pooled, const float* __restrict__ W,
                            const float* __restrict__ b, float* __restrict__ out, int nG) {
  int g = blockIdx.x * blockDim.x + threadIdx.x;
  if (g >= nG) return;
  float l0 = b[0], l1 = b[1];
#pragma unroll
  for (int j = 0; j < 32; ++j) {
    float v = pool_dec(pooled[(size_t)g * 32 + j]);
    if (v == -INFINITY) v = 0.0f;  // seg_max0 empty-segment rule
    l0 += v * W[j * 2 + 0];
    l1 += v * W[j * 2 + 1];
  }
  float mx = fmaxf(l0, l1);
  float lz = mx + logf(expf(l0 - mx) + expf(l1 - mx));
  out[g * 2 + 0] = l0 - lz;
  out[g * 2 + 1] = l1 - lz;
}

// ---------------------------------------------------------------------------
// host orchestration
// ---------------------------------------------------------------------------
extern "C" void kernel_launch(void* const* d_in, const int* in_sizes, int n_in,
                              void* d_out, int out_size, void* d_ws, size_t ws_size,
                              hipStream_t stream) {
  (void)ws_size;
  const float *x_locs = nullptr, *pos = nullptr, *x_cl = nullptr;
  const int *ell = nullptr, *lcs = nullptr, *lcd = nullptr, *ecc = nullptr, *batch = nullptr;
  const float *mW[9], *mB[9], *bnG[6], *bnB[6];
  const float *sWl[3], *sBl[3], *sWr[3];
  const float *hW = nullptr, *hB = nullptr;
  long nLocs = 1000000, nCl = 50000, nEll = 8000000, nLc = 1000000, nEcc = 800000;
  int  nG = out_size / 2;

  auto F = [&](int i) { return (const float*)d_in[i]; };
  auto I = [&](int i) { return (const int*)d_in[i]; };

  if (n_in >= 20) {
    // pytree flattened into individual leaves
    if (in_sizes[0] == 50000) {
      // whole input dict key-sorted:
      // batch, edge_cc, edge_lc_dst, edge_lc_src, edge_ll, params(sorted), pos, x_clusters, x_locs
      batch = I(0); ecc = I(1); lcd = I(2); lcs = I(3); ell = I(4);
      hW = F(5); hB = F(6);
      for (int m = 0; m < 3; ++m) {
        int base = 7 + m * 10;
        for (int i = 0; i < 2; ++i) { bnB[m * 2 + i] = F(base + 2 * i); bnG[m * 2 + i] = F(base + 2 * i + 1); }
        for (int l = 0; l < 3; ++l) { mW[m * 3 + l] = F(base + 4 + 2 * l); mB[m * 3 + l] = F(base + 5 + 2 * l); }
      }
      for (int s = 0; s < 3; ++s) { sWl[s] = F(37 + 3 * s); sBl[s] = F(38 + 3 * s); sWr[s] = F(39 + 3 * s); }
      pos = F(n_in - 3); x_cl = F(n_in - 2); x_locs = F(n_in - 1);
      nEcc = in_sizes[1] / 2; nLc = in_sizes[2];
      nEll = in_sizes[4] / 2; nCl = in_sizes[n_in - 2] / 9; nLocs = (long)in_sizes[n_in - 1] / 2;
    } else {
      x_locs = F(0); pos = F(1); x_cl = F(2);
      if (in_sizes[3] == 16) {
        // insertion order: mlp{layers W,b x3; bn g,b x2} x3, sage{Wl,bl,Wr}x3, head{W,b}
        for (int m = 0; m < 3; ++m) {
          int base = 3 + m * 10;
          for (int l = 0; l < 3; ++l) { mW[m * 3 + l] = F(base + 2 * l); mB[m * 3 + l] = F(base + 2 * l + 1); }
          for (int i = 0; i < 2; ++i) { bnG[m * 2 + i] = F(base + 6 + 2 * i); bnB[m * 2 + i] = F(base + 7 + 2 * i); }
        }
        for (int s = 0; s < 3; ++s) { sWl[s] = F(33 + 3 * s); sBl[s] = F(34 + 3 * s); sWr[s] = F(35 + 3 * s); }
        hW = F(42); hB = F(43);
      } else {
        // params key-sorted: head{W,b}, mlp{bn b,g x2; layers W,b x3} x3, sage
        hW = F(3); hB = F(4);
        for (int m = 0; m < 3; ++m) {
          int base = 5 + m * 10;
          for (int i = 0; i < 2; ++i) { bnB[m * 2 + i] = F(base + 2 * i); bnG[m * 2 + i] = F(base + 2 * i + 1); }
          for (int l = 0; l < 3; ++l) { mW[m * 3 + l] = F(base + 4 + 2 * l); mB[m * 3 + l] = F(base + 5 + 2 * l); }
        }
        for (int s = 0; s < 3; ++s) { sWl[s] = F(35 + 3 * s); sBl[s] = F(36 + 3 * s); sWr[s] = F(37 + 3 * s); }
      }
      ell = I(n_in - 5); lcs = I(n_in - 4); lcd = I(n_in - 3); ecc = I(n_in - 2); batch = I(n_in - 1);
      nLocs = (long)in_sizes[0] / 2; nCl = in_sizes[2] / 9;
      nEll = in_sizes[n_in - 5] / 2; nLc = in_sizes[n_in - 3]; nEcc = in_sizes[n_in - 2] / 2;
    }
  } else {
    // params concatenated into a single flat buffer (key-sorted leaf order)
    x_locs = F(0); pos = F(1); x_cl = F(2);
    const float* p = F(3);
    size_t o = 0;
    hW = p + o; o += 64; hB = p + o; o += 2;
    const int dims[3] = {4, 6, 8};
    for (int m = 0; m < 3; ++m) {
      int D = dims[m];
      for (int i = 0; i < 2; ++i) { bnB[m * 2 + i] = p + o; o += D; bnG[m * 2 + i] = p + o; o += D; }
      for (int l = 0; l < 3; ++l) { mW[m * 3 + l] = p + o; o += D * D; mB[m * 3 + l] = p + o; o += D; }
    }
    const int sin[3] = {17, 18, 24}, sout[3] = {18, 24, 32};
    for (int s = 0; s < 3; ++s) {
      sWl[s] = p + o; o += sin[s] * sout[s];
      sBl[s] = p + o; o += sout[s];
      sWr[s] = p + o; o += sin[s] * sout[s];
    }
    ell = I(4); lcs = I(5); lcd = I(6); ecc = I(7); batch = I(8);
    nLocs = (long)in_sizes[0] / 2; nCl = in_sizes[2] / 9;
    nEll = in_sizes[4] / 2; nLc = in_sizes[6]; nEcc = in_sizes[7] / 2;
  }

  const int* eccS = ecc;
  const int* eccT = ecc + nEcc;

  // ---- workspace layout (floats, 64-elt aligned) ----
  float* wsf = (float*)d_ws;
  size_t off = 0;
  auto alloc = [&](size_t n) { float* q = wsf + off; off += (n + 63) & ~(size_t)63; return q; };
  float*    fold   = alloc(402);
  float*    bufA   = alloc((size_t)nLocs * 8);   // x1 (d=4) then x3 (d=8)
  float*    bufB   = alloc((size_t)nLocs * 6);   // x2 (d=6)
  float*    cagg   = alloc((size_t)nCl * 8);
  float*    xcA    = alloc((size_t)nCl * 32);    // xc0 (17) / xc2 (24)
  float*    xcB    = alloc((size_t)nCl * 32);    // xc1 (18) / xc3 (32)
  float*    sum    = alloc((size_t)nCl * 24);
  float*    cnt    = alloc((size_t)nCl);
  unsigned* pooled = (unsigned*)alloc((size_t)nG * 32);

  auto fill = [&](void* q, unsigned v, long n) {
    fill_u32_kernel<<<(unsigned)((n + 255) / 256), 256, 0, stream>>>((unsigned*)q, v, n);
  };
  auto grid = [](long n, int t) { return (unsigned)((n + t - 1) / t); };

  // ---- fold BN into MLP weights ----
  FoldArgs fa;
  for (int i = 0; i < 9; ++i) { fa.W[i] = mW[i]; fa.B[i] = mB[i]; }
  for (int i = 0; i < 6; ++i) { fa.G[i] = bnG[i]; fa.BNB[i] = bnB[i]; }
  fa.out = fold;
  fold_kernel<<<1, 16, 0, stream>>>(fa);

  // ---- three PointNetConv layers (edge MLP + scatter max) ----
  fill(bufA, 0u, nLocs * 4);
  edge_mlp_kernel<2><<<grid(nEll, 256), 256, 0, stream>>>(ell, ell + nEll, pos, x_locs, fold + 0, bufA, (int)nEll);
  fill(bufB, 0u, nLocs * 6);
  edge_mlp_kernel<4><<<grid(nEll, 256), 256, 0, stream>>>(ell, ell + nEll, pos, bufA, fold + 60, bufB, (int)nEll);
  fill(bufA, 0u, nLocs * 8);
  edge_mlp_kernel<6><<<grid(nEll, 256), 256, 0, stream>>>(ell, ell + nEll, pos, bufB, fold + 186, bufA, (int)nEll);

  // ---- Loc2Cluster max aggregation + concat ----
  fill(cagg, 0u, nCl * 8);
  cagg_kernel<<<grid(nLc, 256), 256, 0, stream>>>(lcs, lcd, bufA, cagg, (int)nLc);
  concat_kernel<<<grid(nCl * 17, 256), 256, 0, stream>>>(x_cl, cagg, xcA, (int)nCl);

  // ---- SAGE mean-aggregation counts (shared across layers) ----
  fill(cnt, 0u, nCl);
  count_kernel<<<grid(nEcc, 256), 256, 0, stream>>>(eccT, cnt, (int)nEcc);

  // ---- three SAGEConv layers (WMMA GEMMs) ----
  unsigned gemmBlocks = grid((nCl + 15) / 16, 4);  // 4 waves/block, 1 tile/wave
  fill(sum, 0u, nCl * 17);
  seg_sum_kernel<<<grid(nEcc, 256), 256, 0, stream>>>(eccS, eccT, xcA, sum, 17, (int)nEcc);
  sage_gemm_kernel<17, 18><<<gemmBlocks, 128, 0, stream>>>(sum, cnt, xcA, sWl[0], sBl[0], sWr[0], xcB, (int)nCl);

  fill(sum, 0u, nCl * 18);
  seg_sum_kernel<<<grid(nEcc, 256), 256, 0, stream>>>(eccS, eccT, xcB, sum, 18, (int)nEcc);
  sage_gemm_kernel<18, 24><<<gemmBlocks, 128, 0, stream>>>(sum, cnt, xcB, sWl[1], sBl[1], sWr[1], xcA, (int)nCl);

  fill(sum, 0u, nCl * 24);
  seg_sum_kernel<<<grid(nEcc, 256), 256, 0, stream>>>(eccS, eccT, xcA, sum, 24, (int)nEcc);
  sage_gemm_kernel<24, 32><<<gemmBlocks, 128, 0, stream>>>(sum, cnt, xcA, sWl[2], sBl[2], sWr[2], xcB, (int)nCl);

  // ---- graph max-pool + head + log_softmax ----
  fill(pooled, POOL_NEG_INF_KEY, (long)nG * 32);
  pool_kernel<<<grid(nCl, 256), 256, 0, stream>>>(batch, xcB, pooled, (int)nCl);
  head_kernel<<<grid(nG, 64), 64, 0, stream>>>(pooled, hW, hB, (float*)d_out, nG);
}
